// ConceptAttentionProto_66520453480505
// MI455X (gfx1250) — matrix operand order repaired and checked by reading, not compiled, not run on hardware
//
#include <hip/hip_runtime.h>
#include <hip/hip_bf16.h>

// ---------------- problem constants ----------------
#define N_IMG 4
#define CHN   512
#define HWSZ  4096
#define FEATD 256
#define POOLN 8192
#define LTILE 128   // spatial pixels per workgroup
#define PCHNK 128   // pool slots per flash chunk

// LDS strides (bf16 elements, padded to dodge bank conflicts, keep 16B align)
#define TH_S 264    // theta_t / agg_t: [128 l][264]  (256 f + pad)
#define PF_S 136    // pool_f:          [256 f][136]  (128 p + pad)
#define PP_S 264    // pool_p:          [128 p][264]  (256 f + pad)
#define WX_S 136    // wexp:            [128 l][136]  (128 p + pad)
#define W1_S 72     // stage1 wth/xb:   [.][72]       (64 k + pad)

#define OFF_THETA 0
#define OFF_POOLF 67584                    // 128*264*2
#define OFF_POOLP 137216                   // + 256*136*2
#define OFF_WEXP  204800                   // + 128*264*2
#define OFF_STATS 239616                   // + 128*136*2
#define SMEM_BYTES 241152                  // + 3*128*4

typedef __attribute__((ext_vector_type(16))) __bf16 bf16x16;
typedef __attribute__((ext_vector_type(8)))  float  f32x8;

union FragU { bf16x16 v; uint4 u[2]; };
union Pack8 { __bf16 h[8]; uint4 u; };

// A-operand frag (16x32 bf16, M x K): lane m = row+(lane&15),
// element j -> K = kBase + 8*(lane>>4) + (j<8 ? j : j+8)
__device__ __forceinline__ bf16x16 ld_fragA(const __bf16* base, int row, int stride,
                                            int kBase, int lane) {
  const __bf16* p = base + (row + (lane & 15)) * stride + kBase + 8 * (lane >> 4);
  FragU f;
  f.u[0] = *reinterpret_cast<const uint4*>(p);
  f.u[1] = *reinterpret_cast<const uint4*>(p + 16);
  return f.v;
}
// B-operand frag (32x16 bf16, K x N) from N-major storage: lane n = row+(lane&15),
// element j -> K = kBase + 8*(lane>>4) + (j<8 ? j : j+8)
__device__ __forceinline__ bf16x16 ld_fragB(const __bf16* base, int row, int stride,
                                            int kBase, int lane) {
  const __bf16* p = base + (row + (lane & 15)) * stride + kBase + 8 * (lane >> 4);
  FragU f;
  f.u[0] = *reinterpret_cast<const uint4*>(p);
  f.u[1] = *reinterpret_cast<const uint4*>(p + 16);
  return f.v;
}
__device__ __forceinline__ uint4 pack8(f32x8 v) {
  Pack8 r;
#pragma unroll
  for (int i = 0; i < 8; ++i) r.h[i] = (__bf16)v[i];
  return r.u;
}
__device__ __forceinline__ Pack8 pack8f(float4 a, float4 b) {
  Pack8 r;
  r.h[0] = (__bf16)a.x; r.h[1] = (__bf16)a.y; r.h[2] = (__bf16)a.z; r.h[3] = (__bf16)a.w;
  r.h[4] = (__bf16)b.x; r.h[5] = (__bf16)b.y; r.h[6] = (__bf16)b.z; r.h[7] = (__bf16)b.w;
  return r;
}
__device__ __forceinline__ f32x8 zero8() {
  f32x8 z;
#pragma unroll
  for (int r = 0; r < 8; ++r) z[r] = 0.0f;
  return z;
}
#define WMMA_BF16(A, B, C) \
  __builtin_amdgcn_wmma_f32_16x16x32_bf16(false, (A), false, (B), (short)0, (C), false, false)

extern "C" __global__ __launch_bounds__(256, 1)
void concept_attn_fused(const float* __restrict__ x,
                        const float* __restrict__ w_theta,
                        const float* __restrict__ w_o,
                        const float* __restrict__ pool,
                        const float* __restrict__ gamma_p,
                        float* __restrict__ out) {
  extern __shared__ char smem[];
  __bf16* theta_t = (__bf16*)(smem + OFF_THETA);  // [128 l][TH_S], A for logits
  __bf16* pool_f  = (__bf16*)(smem + OFF_POOLF);  // [256 f][PF_S], A for agg
  __bf16* pool_p  = (__bf16*)(smem + OFF_POOLP);  // [128 p][PP_S], B for logits
  __bf16* wexp    = (__bf16*)(smem + OFF_WEXP);   // [128 l][WX_S], B for agg
  float*  row_max = (float*)(smem + OFF_STATS);
  float*  row_sum = row_max + 128;
  float*  row_scl = row_sum + 128;
  __bf16* wth   = pool_f;    // stage-1 alias: [256 f][W1_S]
  __bf16* xb    = pool_p;    // stage-1 alias: [128 l][W1_S]
  __bf16* wo    = pool_f;    // stage-3 alias: [128 c][TH_S]
  __bf16* agg_t = theta_t;   // stage-3 B operand: [128 l][TH_S]

  const int tid  = threadIdx.x;
  const int lane = tid & 31;
  const int wv   = tid >> 5;       // wave 0..7
  const int half = lane >> 4;
  const int ln   = lane & 15;
  const int blk  = blockIdx.x;     // 128 blocks: 4 images * 32 l-tiles
  const int n    = blk >> 5;
  const int lBase = (blk & 31) * LTILE;
  const float gamma = gamma_p[0];

  f32x8 acc[16];                   // wave's 16 C tiles: [ft(2) x lt(8)]
#pragma unroll
  for (int i = 0; i < 16; ++i) acc[i] = zero8();

  // ================= Stage 1: theta[256x128] = w_theta * x_tile =================
#pragma unroll 1
  for (int kc = 0; kc < CHN / 64; ++kc) {
    __syncthreads();
    {  // w_theta chunk [256 f][64 k] -> wth (f-major), one row per thread
      const float* src = w_theta + (size_t)tid * CHN + kc * 64;
      __bf16* dst = wth + tid * W1_S;
#pragma unroll
      for (int j = 0; j < 64; j += 8) {
        float4 a = *reinterpret_cast<const float4*>(src + j);
        float4 b = *reinterpret_cast<const float4*>(src + j + 4);
        *reinterpret_cast<uint4*>(dst + j) = pack8f(a, b).u;
      }
    }
    {  // x chunk [64 c][128 l] -> xb transposed (l-major)
      int cl = tid >> 2;
      int ls = (tid & 3) * 32;
      const float* src = x + ((size_t)n * CHN + (size_t)kc * 64 + cl) * HWSZ + lBase + ls;
#pragma unroll
      for (int j = 0; j < 32; j += 4) {
        float4 a = *reinterpret_cast<const float4*>(src + j);
        xb[(ls + j + 0) * W1_S + cl] = (__bf16)a.x;
        xb[(ls + j + 1) * W1_S + cl] = (__bf16)a.y;
        xb[(ls + j + 2) * W1_S + cl] = (__bf16)a.z;
        xb[(ls + j + 3) * W1_S + cl] = (__bf16)a.w;
      }
    }
    __syncthreads();
    // prefetch next K-chunk of w_theta and x while this chunk's WMMAs run
    if (kc + 1 < CHN / 64) {
      const char* pw = (const char*)(w_theta + (size_t)tid * CHN + (kc + 1) * 64);
      __builtin_prefetch(pw, 0, 3);
      __builtin_prefetch(pw + 128, 0, 3);
      int cl = tid >> 2, ls = (tid & 3) * 32;
      const char* px = (const char*)(x + ((size_t)n * CHN + (size_t)(kc + 1) * 64 + cl) * HWSZ +
                                     lBase + ls);
      __builtin_prefetch(px, 0, 3);
    }
#pragma unroll
    for (int ks = 0; ks < 2; ++ks) {
      bf16x16 bfr[8];
#pragma unroll
      for (int lt = 0; lt < 8; ++lt) bfr[lt] = ld_fragB(xb, lt * 16, W1_S, ks * 32, lane);
#pragma unroll
      for (int ft = 0; ft < 2; ++ft) {
        bf16x16 a = ld_fragA(wth, (2 * wv + ft) * 16, W1_S, ks * 32, lane);
#pragma unroll
        for (int lt = 0; lt < 8; ++lt)
          acc[ft * 8 + lt] = WMMA_BF16(a, bfr[lt], acc[ft * 8 + lt]);
      }
    }
  }
  // write theta tiles (C frag: m=f rows contiguous per lane -> one b128 store each)
#pragma unroll
  for (int ft = 0; ft < 2; ++ft)
#pragma unroll
    for (int lt = 0; lt < 8; ++lt) {
      int l  = lt * 16 + ln;
      int fb = (2 * wv + ft) * 16 + 8 * half;
      *reinterpret_cast<uint4*>(theta_t + l * TH_S + fb) = pack8(acc[ft * 8 + lt]);
    }
  if (tid < 128) { row_max[tid] = -3.4e38f; row_sum[tid] = 0.0f; row_scl[tid] = 0.0f; }
#pragma unroll
  for (int i = 0; i < 16; ++i) acc[i] = zero8();   // now agg accumulators

  // ================= Stage 2: flash loop over 8192 pool slots =================
#pragma unroll 1
  for (int pcId = 0; pcId < POOLN / PCHNK; ++pcId) {
    const int pBase = pcId * PCHNK;
    __syncthreads();  // previous chunk's agg reads done before overwrite
    {  // pool chunk [256 f][128 p] -> pool_f (f-major) and pool_p (p-major)
      const float* src = pool + (size_t)tid * POOLN + pBase;
      __bf16* dstF = pool_f + tid * PF_S;
#pragma unroll
      for (int j = 0; j < 128; j += 8) {
        float4 a = *reinterpret_cast<const float4*>(src + j);
        float4 b = *reinterpret_cast<const float4*>(src + j + 4);
        Pack8 r = pack8f(a, b);
        *reinterpret_cast<uint4*>(dstF + j) = r.u;
#pragma unroll
        for (int e = 0; e < 8; ++e) pool_p[(j + e) * PP_S + tid] = r.h[e];
      }
    }
    __syncthreads();
    // prefetch next pool chunk (4 x 128B lines per thread-row) so next
    // iteration's global loads hit cache while this chunk's WMMAs run
    if (pcId + 1 < POOLN / PCHNK) {
      const char* nxt = (const char*)(pool + (size_t)tid * POOLN + pBase + PCHNK);
#pragma unroll
      for (int j = 0; j < 4; ++j) __builtin_prefetch(nxt + j * 128, 0, 3);
    }
    // ---- logits: wave wv owns l-tile wv (16 rows), all 8 p-tiles; K = 256 ----
    f32x8 lg[8];
#pragma unroll
    for (int pt = 0; pt < 8; ++pt) lg[pt] = zero8();
#pragma unroll
    for (int kf = 0; kf < 8; ++kf) {
      bf16x16 a = ld_fragA(theta_t, wv * 16, TH_S, kf * 32, lane);
#pragma unroll
      for (int pt = 0; pt < 8; ++pt) {
        bf16x16 bb = ld_fragB(pool_p, pt * 16, PP_S, kf * 32, lane);
        lg[pt] = WMMA_BF16(a, bb, lg[pt]);
      }
    }
    // ---- online softmax over this chunk (rows live in 16-lane halves) ----
    float mx[8];
#pragma unroll
    for (int r = 0; r < 8; ++r) {
      float m = lg[0][r];
#pragma unroll
      for (int pt = 1; pt < 8; ++pt) m = fmaxf(m, lg[pt][r]);
      mx[r] = m;
    }
#pragma unroll
    for (int off = 8; off >= 1; off >>= 1)
#pragma unroll
      for (int r = 0; r < 8; ++r) mx[r] = fmaxf(mx[r], __shfl_xor(mx[r], off, 32));
    float mn[8], sc[8], sm[8];
#pragma unroll
    for (int r = 0; r < 8; ++r) {
      int lr = wv * 16 + 8 * half + r;
      float mo = row_max[lr];
      mn[r] = fmaxf(mo, mx[r]);
      sc[r] = __expf(mo - mn[r]);
      sm[r] = 0.0f;
    }
#pragma unroll
    for (int pt = 0; pt < 8; ++pt)
#pragma unroll
      for (int r = 0; r < 8; ++r) {
        float e = __expf(lg[pt][r] - mn[r]);
        lg[pt][r] = e;
        sm[r] += e;
      }
#pragma unroll
    for (int off = 8; off >= 1; off >>= 1)
#pragma unroll
      for (int r = 0; r < 8; ++r) sm[r] += __shfl_xor(sm[r], off, 32);
    if (ln == 0) {
#pragma unroll
      for (int r = 0; r < 8; ++r) {
        int lr = wv * 16 + 8 * half + r;
        row_max[lr] = mn[r];
        row_sum[lr] = row_sum[lr] * sc[r] + sm[r];
        row_scl[lr] = sc[r];
      }
    }
    // exp-weights -> wexp[l][p] (B operand for agg, l-major)
#pragma unroll
    for (int pt = 0; pt < 8; ++pt) {
      int p = pt * 16 + ln;
#pragma unroll
      for (int r = 0; r < 8; ++r)
        wexp[(wv * 16 + 8 * half + r) * WX_S + p] = (__bf16)lg[pt][r];
    }
    __syncthreads();
    // ---- rescale running agg, then agg += pool_f * wexp (K = 128) ----
    float scl[8];
#pragma unroll
    for (int lt = 0; lt < 8; ++lt) scl[lt] = row_scl[lt * 16 + ln];
#pragma unroll
    for (int i = 0; i < 16; ++i) {
      float s2 = scl[i & 7];
#pragma unroll
      for (int r = 0; r < 8; ++r) acc[i][r] *= s2;
    }
#pragma unroll
    for (int ks = 0; ks < 4; ++ks) {
      bf16x16 a0 = ld_fragA(pool_f, (2 * wv + 0) * 16, PF_S, ks * 32, lane);
      bf16x16 a1 = ld_fragA(pool_f, (2 * wv + 1) * 16, PF_S, ks * 32, lane);
#pragma unroll
      for (int lt = 0; lt < 8; ++lt) {
        bf16x16 bb = ld_fragB(wexp, lt * 16, WX_S, ks * 32, lane);
        acc[lt]     = WMMA_BF16(a0, bb, acc[lt]);
        acc[8 + lt] = WMMA_BF16(a1, bb, acc[8 + lt]);
      }
    }
  }

  // ============ finalize: agg /= row_sum, store bf16 into agg_t[l][f] ============
  __syncthreads();
  float inv[8];
#pragma unroll
  for (int lt = 0; lt < 8; ++lt) inv[lt] = 1.0f / row_sum[lt * 16 + ln];
#pragma unroll
  for (int ft = 0; ft < 2; ++ft)
#pragma unroll
    for (int lt = 0; lt < 8; ++lt) {
      f32x8 v = acc[ft * 8 + lt];
#pragma unroll
      for (int r = 0; r < 8; ++r) v[r] *= inv[lt];
      int l  = lt * 16 + ln;
      int fb = (2 * wv + ft) * 16 + 8 * half;
      *reinterpret_cast<uint4*>(agg_t + l * TH_S + fb) = pack8(v);
    }

  // ================= Stage 3: o = w_o * agg, out = x + gamma*o =================
#pragma unroll 1
  for (int cg = 0; cg < 4; ++cg) {
    __syncthreads();
    {  // w_o c-group chunk [128 c][256 f] -> wo (c-major)
      int cl = tid >> 1;
      int fs = (tid & 1) * 128;
      const float* src = w_o + ((size_t)cg * 128 + cl) * FEATD + fs;
      __bf16* dst = wo + cl * TH_S + fs;
#pragma unroll
      for (int j = 0; j < 128; j += 8) {
        float4 a = *reinterpret_cast<const float4*>(src + j);
        float4 b = *reinterpret_cast<const float4*>(src + j + 4);
        *reinterpret_cast<uint4*>(dst + j) = pack8f(a, b).u;
      }
    }
    __syncthreads();
    // prefetch next w_o c-group while this group's WMMAs run
    if (cg + 1 < 4) {
      int cl = tid >> 1, fs = (tid & 1) * 128;
      const char* pw = (const char*)(w_o + ((size_t)(cg + 1) * 128 + cl) * FEATD + fs);
      __builtin_prefetch(pw, 0, 3);
      __builtin_prefetch(pw + 128, 0, 3);
      __builtin_prefetch(pw + 256, 0, 3);
      __builtin_prefetch(pw + 384, 0, 3);
    }
    f32x8 oc[8];
#pragma unroll
    for (int lt = 0; lt < 8; ++lt) oc[lt] = zero8();
#pragma unroll
    for (int kf = 0; kf < 8; ++kf) {
      bf16x16 a = ld_fragA(wo, wv * 16, TH_S, kf * 32, lane);
#pragma unroll
      for (int lt = 0; lt < 8; ++lt) {
        bf16x16 bb = ld_fragB(agg_t, lt * 16, TH_S, kf * 32, lane);
        oc[lt] = WMMA_BF16(a, bb, oc[lt]);
      }
    }
#pragma unroll
    for (int lt = 0; lt < 8; ++lt) {
      int l = lBase + lt * 16 + ln;
#pragma unroll
      for (int r = 0; r < 8; ++r) {
        int c = cg * 128 + wv * 16 + 8 * half + r;
        size_t idx = ((size_t)n * CHN + c) * (size_t)HWSZ + l;
        out[idx] = x[idx] + gamma * oc[lt][r];
      }
    }
  }
}

extern "C" void kernel_launch(void* const* d_in, const int* in_sizes, int n_in,
                              void* d_out, int out_size, void* d_ws, size_t ws_size,
                              hipStream_t stream) {
  const float* x       = (const float*)d_in[0];
  const float* w_theta = (const float*)d_in[1];
  const float* w_o     = (const float*)d_in[2];
  const float* pool    = (const float*)d_in[3];
  const float* gamma   = (const float*)d_in[4];
  float* out = (float*)d_out;

  (void)in_sizes; (void)n_in; (void)out_size; (void)d_ws; (void)ws_size;

  // Opt in to >64KB dynamic LDS (no-op on stacks that don't require it).
  (void)hipFuncSetAttribute(reinterpret_cast<const void*>(concept_attn_fused),
                            hipFuncAttributeMaxDynamicSharedMemorySize, SMEM_BYTES);

  concept_attn_fused<<<dim3(N_IMG * (HWSZ / LTILE)), dim3(256), SMEM_BYTES, stream>>>(
      x, w_theta, w_o, pool, gamma, out);
}
